// MambaBlock_42734924595844
// MI455X (gfx1250) — matrix operand order repaired and checked by reading, not compiled, not run on hardware
//
#include <hip/hip_runtime.h>

// ---------------------------------------------------------------------------
// Mamba block for MI455X (gfx1250, wave32, WMMA).
//  - Dense projections: f16 WMMA (v_wmma_f32_16x16x32_f16), f32 accum,
//    software-pipelined k-loop (next fragments in flight behind WMMAs),
//    guard-free tiles so EXEC stays all-1s around every WMMA.
//  - Selective scan: 3-pass chunked parallel scan (16 chunks x 256 steps),
//    B/C staged in LDS, chunk transfer operator = exp(A * sum(dt)).
// ---------------------------------------------------------------------------

typedef __attribute__((ext_vector_type(8)))  _Float16 v8h;
typedef __attribute__((ext_vector_type(16))) _Float16 v16h;
typedef __attribute__((ext_vector_type(8)))  float    v8f;

#define BATCH   2
#define SEQ     4096
#define DMODEL  1024
#define DINNER  2048
#define DSTATE  16
#define DTRANK  64
#define NCHUNK  16
#define CHUNKL  256      // SEQ / NCHUNK

#define WMMA_F16(Af, Bf, Cf) \
  __builtin_amdgcn_wmma_f32_16x16x32_f16(false, (Af), false, (Bf), (short)0, (Cf), false, false)

// ---------------------------------------------------------------------------
// f32 -> f16 conversion
// ---------------------------------------------------------------------------
__global__ void f32_to_f16_kernel(const float* __restrict__ src,
                                  _Float16* __restrict__ dst, size_t n) {
  size_t i = (size_t)blockIdx.x * blockDim.x + threadIdx.x;
  if (i < n) dst[i] = (_Float16)src[i];
}

// ---------------------------------------------------------------------------
// Fragment loads.
// A frag (16x32 MxK): lane = row m, contiguous k 0-15 | 16-31.
// B frag (32x16 KxN): lane = col n, contiguous k 0-15 | 16-31 (row-major W).
// ---------------------------------------------------------------------------
__device__ __forceinline__ v16h ld_a(const _Float16* __restrict__ p) {
  v8h lo = *(const v8h*)(p);
  v8h hi = *(const v8h*)(p + 8);
  return __builtin_shufflevector(lo, hi, 0,1,2,3,4,5,6,7,8,9,10,11,12,13,14,15);
}
__device__ __forceinline__ v16h ld_b(const _Float16* __restrict__ p) {
  v8h lo = *(const v8h*)(p);
  v8h hi = *(const v8h*)(p + 8);
  return __builtin_shufflevector(lo, hi, 0,1,2,3,4,5,6,7,8,9,10,11,12,13,14,15);
}

// ---------------------------------------------------------------------------
// C[M,N] = A[M,K] @ W[N,K]^T, N multiple of 256. block=(32,4): each wave owns
// a 16x64 tile, exact grid cover -> no guards, EXEC all-1s throughout.
// ---------------------------------------------------------------------------
__global__ void wmma_gemm_n64(const _Float16* __restrict__ A,
                              const _Float16* __restrict__ W,
                              float* __restrict__ C,
                              int K, int lda, int ldb, int ldc) {
  const int lane = threadIdx.x;
  const int row0 = blockIdx.x * 16;
  const int colb = (blockIdx.y * 4 + threadIdx.y) * 64;
  const int r  = lane & 15;
  const int hi = lane >> 4;

  const _Float16* pa  = A + (size_t)(row0 + r) * lda + (hi ? 16 : 0);
  const _Float16* pb0 = W + (size_t)(colb + r) * ldb + (hi ? 16 : 0);
  const _Float16* pb1 = pb0 + (size_t)16 * ldb;
  const _Float16* pb2 = pb0 + (size_t)32 * ldb;
  const _Float16* pb3 = pb0 + (size_t)48 * ldb;

  v8f c0 = {}, c1 = {}, c2 = {}, c3 = {};
  v16h a  = ld_a(pa);
  v16h b0 = ld_b(pb0), b1 = ld_b(pb1), b2 = ld_b(pb2), b3 = ld_b(pb3);
  for (int k0 = 32; k0 < K; k0 += 32) {
    pa += 32; pb0 += 32; pb1 += 32; pb2 += 32; pb3 += 32;
    // prefetch next k-slice while WMMAs for current slice execute
    v16h an = ld_a(pa);
    v16h n0 = ld_b(pb0), n1 = ld_b(pb1), n2 = ld_b(pb2), n3 = ld_b(pb3);
    c0 = WMMA_F16(a, b0, c0);
    c1 = WMMA_F16(a, b1, c1);
    c2 = WMMA_F16(a, b2, c2);
    c3 = WMMA_F16(a, b3, c3);
    a = an; b0 = n0; b1 = n1; b2 = n2; b3 = n3;
  }
  c0 = WMMA_F16(a, b0, c0);
  c1 = WMMA_F16(a, b1, c1);
  c2 = WMMA_F16(a, b2, c2);
  c3 = WMMA_F16(a, b3, c3);

  // D layout: lane r -> col n=r, VGPR i -> row (lane<16 ? i : 8+i)
  const int mrow = row0 + (hi ? 8 : 0);
  float* cp = C + (size_t)mrow * ldc + colb + r;
  #pragma unroll
  for (int i = 0; i < 8; ++i) cp[(size_t)i * ldc + 0]  = c0[i];
  #pragma unroll
  for (int i = 0; i < 8; ++i) cp[(size_t)i * ldc + 16] = c1[i];
  #pragma unroll
  for (int i = 0; i < 8; ++i) cp[(size_t)i * ldc + 32] = c2[i];
  #pragma unroll
  for (int i = 0; i < 8; ++i) cp[(size_t)i * ldc + 48] = c3[i];
}

// ---------------------------------------------------------------------------
// C[M,16] = A[M,K] @ W[16,K]^T  (Bm/Cm projections). block=(32,4): each wave
// owns its own 16-row block; grid.x = M/64. No guards.
// ---------------------------------------------------------------------------
__global__ void wmma_gemm_n16(const _Float16* __restrict__ A,
                              const _Float16* __restrict__ W,
                              float* __restrict__ C,
                              int K, int lda, int ldb, int ldc) {
  const int lane = threadIdx.x;
  const int row0 = (blockIdx.x * 4 + threadIdx.y) * 16;
  const int r  = lane & 15;
  const int hi = lane >> 4;

  const _Float16* pa = A + (size_t)(row0 + r) * lda + (hi ? 16 : 0);
  const _Float16* pb = W + (size_t)r * ldb + (hi ? 16 : 0);

  v8f c0 = {};
  v16h a = ld_a(pa);
  v16h b = ld_b(pb);
  for (int k0 = 32; k0 < K; k0 += 32) {
    pa += 32; pb += 32;
    v16h an = ld_a(pa);
    v16h bn = ld_b(pb);
    c0 = WMMA_F16(a, b, c0);
    a = an; b = bn;
  }
  c0 = WMMA_F16(a, b, c0);

  const int mrow = row0 + (hi ? 8 : 0);
  float* cp = C + (size_t)mrow * ldc + r;
  #pragma unroll
  for (int i = 0; i < 8; ++i) cp[(size_t)i * ldc] = c0[i];
}

// ---------------------------------------------------------------------------
// Depthwise causal conv1d(4) + bias + SiLU.  Reads x_val_raw (xz[:, :Di]),
// neighbor taps hit L2.
// ---------------------------------------------------------------------------
__global__ void conv_silu_kernel(const float* __restrict__ xz,
                                 const float* __restrict__ cw,
                                 const float* __restrict__ cb,
                                 float* __restrict__ xval) {
  size_t i = (size_t)blockIdx.x * blockDim.x + threadIdx.x;
  if (i >= (size_t)BATCH * SEQ * DINNER) return;
  const int d = (int)(i % DINNER);
  const size_t t = i / DINNER;              // b*SEQ + l
  const int l = (int)(t % SEQ);
  float acc = cb[d];
  const float* col = xz + t * (size_t)(2 * DINNER) + d;
  #pragma unroll
  for (int j = 0; j < 4; ++j) {
    const int ll = l - 3 + j;
    if (ll >= 0)
      acc = fmaf(cw[d * 4 + j], col[(long)(j - 3) * (2 * DINNER)], acc);
  }
  xval[i] = acc / (1.0f + __expf(-acc));    // silu
}

// dt = softplus(dt_raw + b_dt), in place
__global__ void dt_softplus_kernel(float* __restrict__ dt,
                                   const float* __restrict__ bdt) {
  size_t i = (size_t)blockIdx.x * blockDim.x + threadIdx.x;
  if (i >= (size_t)BATCH * SEQ * DINNER) return;
  const int d = (int)(i % DINNER);
  const float v = dt[i] + bdt[d];
  dt[i] = (v > 20.0f) ? v : log1pf(__expf(v));
}

// ---------------------------------------------------------------------------
// Scan pass 1: per-chunk local scan with h_in = 0.
// Writes local y over xval (in place), stores chunk operator P = exp(A*sum dt)
// and local end-state S.  B/C for the chunk staged in LDS (32 KB).
// ---------------------------------------------------------------------------
__global__ void scan_pass1(const float* __restrict__ dt,
                           float* __restrict__ xval_y,
                           const float* __restrict__ Bm,
                           const float* __restrict__ Cm,
                           const float* __restrict__ A_log,
                           float* __restrict__ P, float* __restrict__ S) {
  __shared__ float sB[CHUNKL * DSTATE];
  __shared__ float sC[CHUNKL * DSTATE];
  const int c = blockIdx.x & (NCHUNK - 1);
  const int b = blockIdx.x >> 4;
  const int d = blockIdx.y * blockDim.x + threadIdx.x;
  const size_t tok0 = (size_t)b * SEQ + (size_t)c * CHUNKL;
  for (int i = threadIdx.x; i < CHUNKL * DSTATE; i += blockDim.x) {
    sB[i] = Bm[tok0 * DSTATE + i];
    sC[i] = Cm[tok0 * DSTATE + i];
  }
  __syncthreads();

  float Ad[DSTATE], h[DSTATE];
  #pragma unroll
  for (int n = 0; n < DSTATE; ++n) {
    Ad[n] = -__expf(A_log[d * DSTATE + n]);
    h[n] = 0.0f;
  }
  float dtsum = 0.0f;
  for (int ll = 0; ll < CHUNKL; ++ll) {
    const size_t idx = (tok0 + ll) * DINNER + d;
    const float dtv = dt[idx];
    const float xv  = xval_y[idx];
    const float du  = dtv * xv;
    dtsum += dtv;
    float y = 0.0f;
    #pragma unroll
    for (int n = 0; n < DSTATE; ++n) {
      const float dA = __expf(dtv * Ad[n]);
      h[n] = fmaf(dA, h[n], du * sB[ll * DSTATE + n]);
      y = fmaf(h[n], sC[ll * DSTATE + n], y);
    }
    xval_y[idx] = y;                        // in-place: this thread owns (l,d)
  }
  const size_t o = (((size_t)b * NCHUNK + c) * DINNER + d) * DSTATE;
  #pragma unroll
  for (int n = 0; n < DSTATE; ++n) {
    P[o + n] = __expf(Ad[n] * dtsum);       // product of dA collapses to 1 exp
    S[o + n] = h[n];
  }
}

// Sequential combine across the 16 chunks (tiny): h_in[c] for every (b,d).
__global__ void scan_fix(const float* __restrict__ P,
                         const float* __restrict__ S,
                         float* __restrict__ Hin) {
  const int t = blockIdx.x * blockDim.x + threadIdx.x;   // over BATCH*DINNER
  const int d = t % DINNER;
  const int b = t / DINNER;
  float h[DSTATE];
  #pragma unroll
  for (int n = 0; n < DSTATE; ++n) h[n] = 0.0f;
  for (int c = 0; c < NCHUNK; ++c) {
    const size_t o = (((size_t)b * NCHUNK + c) * DINNER + d) * DSTATE;
    #pragma unroll
    for (int n = 0; n < DSTATE; ++n) {
      Hin[o + n] = h[n];
      h[n] = fmaf(P[o + n], h[n], S[o + n]);
    }
  }
}

// Pass 3: y[l] += sum_n C[l,n] * exp(A_n * prefix_sum(dt)) * h_in[n]
__global__ void scan_pass3(const float* __restrict__ dt,
                           float* __restrict__ y,
                           const float* __restrict__ Cm,
                           const float* __restrict__ A_log,
                           const float* __restrict__ Hin) {
  __shared__ float sC[CHUNKL * DSTATE];
  const int c = blockIdx.x & (NCHUNK - 1);
  const int b = blockIdx.x >> 4;
  if (c == 0) return;                        // h_in == 0, no correction
  const int d = blockIdx.y * blockDim.x + threadIdx.x;
  const size_t tok0 = (size_t)b * SEQ + (size_t)c * CHUNKL;
  for (int i = threadIdx.x; i < CHUNKL * DSTATE; i += blockDim.x)
    sC[i] = Cm[tok0 * DSTATE + i];
  __syncthreads();

  float Ad[DSTATE], hi[DSTATE];
  const size_t o = (((size_t)b * NCHUNK + c) * DINNER + d) * DSTATE;
  #pragma unroll
  for (int n = 0; n < DSTATE; ++n) {
    Ad[n] = -__expf(A_log[d * DSTATE + n]);
    hi[n] = Hin[o + n];
  }
  float dts = 0.0f;
  for (int ll = 0; ll < CHUNKL; ++ll) {
    const size_t idx = (tok0 + ll) * DINNER + d;
    dts += dt[idx];                          // include dt[l] before correction
    float corr = 0.0f;
    #pragma unroll
    for (int n = 0; n < DSTATE; ++n)
      corr = fmaf(sC[ll * DSTATE + n] * hi[n], __expf(Ad[n] * dts), corr);
    y[idx] += corr;
  }
}

// Gate y *= silu(z) and convert to f16 for the output WMMA GEMM.
__global__ void gate_to_f16(const float* __restrict__ y,
                            const float* __restrict__ xz,
                            _Float16* __restrict__ yh) {
  size_t i = (size_t)blockIdx.x * blockDim.x + threadIdx.x;
  if (i >= (size_t)BATCH * SEQ * DINNER) return;
  const int d = (int)(i % DINNER);
  const size_t t = i / DINNER;
  const float z = xz[t * (size_t)(2 * DINNER) + DINNER + d];
  const float g = y[i] * (z / (1.0f + __expf(-z)));
  yh[i] = (_Float16)g;
}

// ---------------------------------------------------------------------------
extern "C" void kernel_launch(void* const* d_in, const int* in_sizes, int n_in,
                              void* d_out, int out_size, void* d_ws, size_t ws_size,
                              hipStream_t stream) {
  (void)in_sizes; (void)n_in; (void)out_size; (void)ws_size;
  const float* x     = (const float*)d_in[0];
  const float* W_in  = (const float*)d_in[1];
  const float* convw = (const float*)d_in[2];
  const float* convb = (const float*)d_in[3];
  const float* A_log = (const float*)d_in[4];
  const float* W_dt  = (const float*)d_in[5];
  const float* b_dt  = (const float*)d_in[6];
  const float* W_B   = (const float*)d_in[7];
  const float* W_C   = (const float*)d_in[8];
  const float* W_out = (const float*)d_in[9];
  float* out = (float*)d_out;

  const size_t M = (size_t)BATCH * SEQ;     // 8192 tokens
  char* ws = (char*)d_ws;
  size_t off = 0;
  auto alloc = [&](size_t bytes) -> char* {
    char* p = ws + off;
    off = (off + bytes + 255) & ~(size_t)255;
    return p;
  };
  _Float16* x_h    = (_Float16*)alloc(M * DMODEL * 2);
  _Float16* Win_h  = (_Float16*)alloc((size_t)2 * DINNER * DMODEL * 2);
  _Float16* Wdt_h  = (_Float16*)alloc((size_t)DINNER * DTRANK * 2);
  _Float16* WB_h   = (_Float16*)alloc((size_t)DSTATE * DMODEL * 2);
  _Float16* WC_h   = (_Float16*)alloc((size_t)DSTATE * DMODEL * 2);
  _Float16* Wout_h = (_Float16*)alloc((size_t)DMODEL * DINNER * 2);
  float* xz   = (float*)alloc(M * (size_t)(2 * DINNER) * 4);
  float* dtb  = (float*)alloc(M * (size_t)DINNER * 4);
  float* BmB  = (float*)alloc(M * (size_t)DSTATE * 4);
  float* CmB  = (float*)alloc(M * (size_t)DSTATE * 4);
  float* xval = (float*)alloc(M * (size_t)DINNER * 4);   // y in place later
  const size_t chunkstate = (size_t)BATCH * NCHUNK * DINNER * DSTATE * 4;
  float* Pb  = (float*)alloc(chunkstate);
  float* Sb  = (float*)alloc(chunkstate);
  float* Hin = (float*)alloc(chunkstate);
  _Float16* yh = (_Float16*)alloc(M * (size_t)DINNER * 2);

  auto cvt = [&](const float* s, _Float16* d, size_t n) {
    f32_to_f16_kernel<<<dim3((unsigned)((n + 255) / 256)), dim3(256), 0, stream>>>(s, d, n);
  };
  cvt(x,     x_h,    M * DMODEL);
  cvt(W_in,  Win_h,  (size_t)2 * DINNER * DMODEL);
  cvt(W_dt,  Wdt_h,  (size_t)DINNER * DTRANK);
  cvt(W_B,   WB_h,   (size_t)DSTATE * DMODEL);
  cvt(W_C,   WC_h,   (size_t)DSTATE * DMODEL);
  cvt(W_out, Wout_h, (size_t)DMODEL * DINNER);

  const dim3 gblk(32, 4, 1);
  // xz = x @ W_in^T   (M x 4096, K=1024)
  wmma_gemm_n64<<<dim3((unsigned)(M / 16), (2 * DINNER) / 256), gblk, 0, stream>>>(
      x_h, Win_h, xz, DMODEL, DMODEL, DMODEL, 2 * DINNER);
  // dt_raw = x[:, :64] @ W_dt^T  (M x 2048, K=64, lda=1024)
  wmma_gemm_n64<<<dim3((unsigned)(M / 16), DINNER / 256), gblk, 0, stream>>>(
      x_h, Wdt_h, dtb, DTRANK, DMODEL, DTRANK, DINNER);
  // Bm, Cm = x @ W_{B,C}^T  (M x 16, K=1024); 4 row-blocks per block
  wmma_gemm_n16<<<dim3((unsigned)(M / 64)), gblk, 0, stream>>>(
      x_h, WB_h, BmB, DMODEL, DMODEL, DMODEL, DSTATE);
  wmma_gemm_n16<<<dim3((unsigned)(M / 64)), gblk, 0, stream>>>(
      x_h, WC_h, CmB, DMODEL, DMODEL, DMODEL, DSTATE);

  const size_t tot = M * DINNER;
  conv_silu_kernel<<<dim3((unsigned)((tot + 255) / 256)), dim3(256), 0, stream>>>(
      xz, convw, convb, xval);
  dt_softplus_kernel<<<dim3((unsigned)((tot + 255) / 256)), dim3(256), 0, stream>>>(
      dtb, b_dt);

  // chunked parallel scan: 32 chunk-blocks x 8 channel-blocks, 256 thr each
  scan_pass1<<<dim3(BATCH * NCHUNK, DINNER / 256), dim3(256), 0, stream>>>(
      dtb, xval, BmB, CmB, A_log, Pb, Sb);
  scan_fix<<<dim3((BATCH * DINNER) / 256), dim3(256), 0, stream>>>(Pb, Sb, Hin);
  scan_pass3<<<dim3(BATCH * NCHUNK, DINNER / 256), dim3(256), 0, stream>>>(
      dtb, xval, CmB, A_log, Hin);

  gate_to_f16<<<dim3((unsigned)((tot + 255) / 256)), dim3(256), 0, stream>>>(
      xval, xz, yh);
  // out = y @ W_out^T  (M x 1024, K=2048)
  wmma_gemm_n64<<<dim3((unsigned)(M / 16), DMODEL / 256), gblk, 0, stream>>>(
      yh, Wout_h, out, DINNER, DINNER, DINNER, DMODEL);
}
